// TargetEncoder_4930622456139
// MI455X (gfx1250) — compile-verified, matrix operand order checked by reading, not run on hardware
//
#include <hip/hip_runtime.h>
#include <hip/hip_bf16.h>

typedef __bf16 bf16_t;
typedef bf16_t v16bf __attribute__((ext_vector_type(16)));
typedef bf16_t v8bf  __attribute__((ext_vector_type(8)));
typedef bf16_t v4bf  __attribute__((ext_vector_type(4)));
typedef float  v8f   __attribute__((ext_vector_type(8)));

#define NNODES 50000

// ---------------- degree / normalization ----------------

__global__ void k_degree(const long long* __restrict__ ei, long long E,
                         float* __restrict__ deg) {
  long long e = (long long)blockIdx.x * blockDim.x + threadIdx.x;
  if (e < E) unsafeAtomicAdd(&deg[ei[E + e]], 1.0f);  // row = dst = ei[1][e]
}

__global__ void k_deg_finalize(const float* __restrict__ deg,
                               float* __restrict__ dis,
                               float* __restrict__ diagw, int n) {
  int i = blockIdx.x * blockDim.x + threadIdx.x;
  if (i >= n) return;
  float d = deg[i];
  bool pos = d > 0.0f;
  dis[i]   = pos ? rsqrtf(d) : 0.0f;
  diagw[i] = pos ? 0.0f : -1.0f;
}

// ---------------- converts / packing ----------------

__global__ void k_f32_to_bf16(const float* __restrict__ src, bf16_t* __restrict__ dst,
                              long long total) {
  long long i = (long long)blockIdx.x * blockDim.x + threadIdx.x;
  if (i < total) dst[i] = (bf16_t)src[i];
}

// Wp[t][n][k] (bf16, zero-padded) from W[t][k][n] (f32)
__global__ void k_pack_w(const float* __restrict__ W, bf16_t* __restrict__ Wp,
                         int Fin, int Fout, int Kpad, int Npad) {
  long long idx = (long long)blockIdx.x * blockDim.x + threadIdx.x;
  long long tot = 3LL * Npad * Kpad;
  if (idx >= tot) return;
  int k = (int)(idx % Kpad);
  long long tn = idx / Kpad;
  int n = (int)(tn % Npad);
  int t = (int)(tn / Npad);
  float v = (k < Fin && n < Fout) ? W[((size_t)t * Fin + k) * Fout + n] : 0.0f;
  Wp[idx] = (bf16_t)v;
}

// ---------------- SpMM scatter: agg[dst] += w_e * h[src] (bf16 gather, f32 atomics) ----

__global__ void k_spmm(const long long* __restrict__ ei, long long E,
                       const float* __restrict__ dis, const bf16_t* __restrict__ h,
                       float* __restrict__ agg, int ldShift) {
  int chunkShift = ldShift - 2;                 // 4 features per thread
  long long idx = (long long)blockIdx.x * blockDim.x + threadIdx.x;
  long long tot = E << chunkShift;
  if (idx >= tot) return;
  long long e = idx >> chunkShift;
  int f0 = ((int)idx & ((1 << chunkShift) - 1)) << 2;
  long long src = ei[e];       // col
  long long dst = ei[E + e];   // row
  float w = -dis[dst] * dis[src];
  const bf16_t* hp = h + (src << ldShift) + f0;
  float* ap = agg + (dst << ldShift) + f0;
  v4bf hv = *(const v4bf*)hp;
  unsafeAtomicAdd(ap + 0, w * (float)hv[0]);
  unsafeAtomicAdd(ap + 1, w * (float)hv[1]);
  unsafeAtomicAdd(ap + 2, w * (float)hv[2]);
  unsafeAtomicAdd(ap + 3, w * (float)hv[3]);
}

// T1_bf = bf16(agg + diag_w * h)
__global__ void k_finalize_T1(const float* __restrict__ agg, const bf16_t* __restrict__ h,
                              const float* __restrict__ diagw, bf16_t* __restrict__ T1,
                              long long total, int ldShift) {
  long long i = (long long)blockIdx.x * blockDim.x + threadIdx.x;
  if (i >= total) return;
  long long r = i >> ldShift;
  T1[i] = (bf16_t)(agg[i] + diagw[r] * (float)h[i]);
}

// T2_bf = bf16(2*(agg + diag_w*T1) - h)
__global__ void k_make_T2(const float* __restrict__ agg, const bf16_t* __restrict__ T1,
                          const bf16_t* __restrict__ h, const float* __restrict__ diagw,
                          bf16_t* __restrict__ T2, long long total, int ldShift) {
  long long i = (long long)blockIdx.x * blockDim.x + threadIdx.x;
  if (i >= total) return;
  long long r = i >> ldShift;
  T2[i] = (bf16_t)(2.0f * (agg[i] + diagw[r] * (float)T1[i]) - (float)h[i]);
}

// ---------------- fused 3-term WMMA GEMM ----------------
// out = relu(T0@W[0] + T1@W[1] + T2@W[2] + b)
// Wave computes a 16-row x 64-col strip: 1 A-tile reused across 4 B-tiles / 4 accumulators.
// All operands bf16 with padded dims -> every fragment load is an unguarded b128.

__device__ __forceinline__ v16bf cat8(v8bf lo, v8bf hi) {
  return __builtin_shufflevector(lo, hi, 0, 1, 2, 3, 4, 5, 6, 7,
                                 8, 9, 10, 11, 12, 13, 14, 15);
}

__global__ void __launch_bounds__(128)
k_wmma_gemm3(const bf16_t* __restrict__ T0, const bf16_t* __restrict__ T1,
             const bf16_t* __restrict__ T2, const bf16_t* __restrict__ Wp,
             const float* __restrict__ bias, bf16_t* __restrict__ out_bf,
             float* __restrict__ out_f32, int M, int Kpad, int Npad, int Fout) {
  int wave = threadIdx.x >> 5;
  int lane = threadIdx.x & 31;
  int m0 = (blockIdx.x * 4 + wave) * 16;
  if (m0 >= M) return;            // uniform per wave; WMMA runs with full EXEC
  int n0 = blockIdx.y * 64;
  int l15 = lane & 15;
  int kofs = (lane < 16) ? 0 : 8; // A/B bf16 fragment K striping per ISA layout

  v8f acc[4];
#pragma unroll
  for (int j = 0; j < 4; ++j) acc[j] = (v8f){0.f,0.f,0.f,0.f,0.f,0.f,0.f,0.f};

  int nkb = Kpad >> 5;
#pragma unroll
  for (int t = 0; t < 3; ++t) {
    const bf16_t* Ap = ((t == 0) ? T0 : (t == 1) ? T1 : T2)
                       + (size_t)(m0 + l15) * Kpad + kofs;
    const bf16_t* Bp = Wp + ((size_t)t * Npad + n0 + l15) * Kpad + kofs;
    for (int kb = 0; kb < nkb; ++kb) {
      v16bf a = cat8(*(const v8bf*)Ap, *(const v8bf*)(Ap + 16));
#pragma unroll
      for (int j = 0; j < 4; ++j) {
        const bf16_t* q = Bp + (size_t)j * 16 * Kpad;
        v16bf b = cat8(*(const v8bf*)q, *(const v8bf*)(q + 16));
        acc[j] = __builtin_amdgcn_wmma_f32_16x16x32_bf16(
            false, a, false, b, (short)0, acc[j], false, false);
      }
      Ap += 32;
      Bp += 32;
    }
  }

  // C/D layout: VGPR r -> row m0+r (lanes 0-15) / m0+8+r (lanes 16-31), col = n0' + l15
  int rb = m0 + ((lane < 16) ? 0 : 8);
#pragma unroll
  for (int j = 0; j < 4; ++j) {
    int col = n0 + j * 16 + l15;
    float bv = (col < Fout) ? bias[col] : 0.0f;
#pragma unroll
    for (int r = 0; r < 8; ++r) {
      float v = fmaxf(acc[j][r] + bv, 0.0f);   // padding cols: acc=0, bv=0 -> stores 0
      size_t row = (size_t)(rb + r);
      if (out_bf) out_bf[row * Npad + col] = (bf16_t)v;
      else if (col < Fout) out_f32[row * Fout + col] = v;
    }
  }
}

// ---------------- host orchestration ----------------

static inline long long cdiv_ll(long long a, long long b) { return (a + b - 1) / b; }

extern "C" void kernel_launch(void* const* d_in, const int* in_sizes, int n_in,
                              void* d_out, int out_size, void* d_ws, size_t ws_size,
                              hipStream_t stream) {
  const float*     x  = (const float*)d_in[0];
  const long long* ei = (const long long*)d_in[1];
  const float* w1 = (const float*)d_in[2];
  const float* b1 = (const float*)d_in[3];
  const float* w2 = (const float*)d_in[4];
  const float* b2 = (const float*)d_in[5];
  const float* w3 = (const float*)d_in[6];
  const float* b3 = (const float*)d_in[7];

  const long long E = in_sizes[1] / 2;
  const int N = NNODES;

  // workspace layout
  char* ws = (char*)d_ws;
  float* deg   = (float*)ws;                 ws += (size_t)N * 4;
  float* dis   = (float*)ws;                 ws += (size_t)N * 4;
  float* diagw = (float*)ws;                 ws += (size_t)N * 4;
  float* aggF  = (float*)ws;                 ws += (size_t)N * 512 * 4;   // shared f32 agg
  bf16_t* x_bf = (bf16_t*)ws;                ws += (size_t)N * 512 * 2;
  bf16_t* T1   = (bf16_t*)ws;                ws += (size_t)N * 512 * 2;
  bf16_t* T2   = (bf16_t*)ws;                ws += (size_t)N * 512 * 2;
  bf16_t* h1   = (bf16_t*)ws;                ws += (size_t)N * 256 * 2;   // ld=256 (Fout=250)
  bf16_t* h2   = (bf16_t*)ws;                ws += (size_t)N * 512 * 2;   // ld=512 (Fout=500)
  bf16_t* Wp1  = (bf16_t*)ws;                ws += 3LL * 256 * 512 * 2;   // [t][256][512]
  bf16_t* Wp2  = (bf16_t*)ws;                ws += 3LL * 512 * 256 * 2;   // [t][512][256]
  bf16_t* Wp3  = (bf16_t*)ws;                ws += 3LL * 1024 * 512 * 2;  // [t][1024][512]

  // degree + normalization
  hipMemsetAsync(deg, 0, (size_t)N * sizeof(float), stream);
  k_degree<<<(unsigned)cdiv_ll(E, 256), 256, 0, stream>>>(ei, E, deg);
  k_deg_finalize<<<(N + 255) / 256, 256, 0, stream>>>(deg, dis, diagw, N);

  // convert x, pack weights (zero-padded, transposed, bf16)
  k_f32_to_bf16<<<(unsigned)cdiv_ll((long long)N * 512, 256), 256, 0, stream>>>(
      x, x_bf, (long long)N * 512);
  k_pack_w<<<(unsigned)cdiv_ll(3LL * 256 * 512, 256), 256, 0, stream>>>(w1, Wp1, 512, 250, 512, 256);
  k_pack_w<<<(unsigned)cdiv_ll(3LL * 512 * 256, 256), 256, 0, stream>>>(w2, Wp2, 250, 500, 256, 512);
  k_pack_w<<<(unsigned)cdiv_ll(3LL * 1024 * 512, 256), 256, 0, stream>>>(w3, Wp3, 500, 1000, 512, 1024);

  auto run_layer = [&](const bf16_t* h, int ldShift, const bf16_t* Wp, const float* bl,
                       bf16_t* o_bf, float* o_f32, int Npad, int Fout) {
    int Kpad = 1 << ldShift;
    long long tot = (long long)N << ldShift;
    long long sp_threads = E << (ldShift - 2);

    // T1 = Lhat(h)
    hipMemsetAsync(aggF, 0, (size_t)tot * sizeof(float), stream);
    k_spmm<<<(unsigned)cdiv_ll(sp_threads, 256), 256, 0, stream>>>(ei, E, dis, h, aggF, ldShift);
    k_finalize_T1<<<(unsigned)cdiv_ll(tot, 256), 256, 0, stream>>>(aggF, h, diagw, T1, tot, ldShift);

    // T2 = 2*Lhat(T1) - h
    hipMemsetAsync(aggF, 0, (size_t)tot * sizeof(float), stream);
    k_spmm<<<(unsigned)cdiv_ll(sp_threads, 256), 256, 0, stream>>>(ei, E, dis, T1, aggF, ldShift);
    k_make_T2<<<(unsigned)cdiv_ll(tot, 256), 256, 0, stream>>>(aggF, T1, h, diagw, T2, tot, ldShift);

    // out = relu(h@W0 + T1@W1 + T2@W2 + b)
    dim3 grid((N / 16 + 3) / 4, Npad / 64);
    k_wmma_gemm3<<<grid, 128, 0, stream>>>(h, T1, T2, Wp, bl, o_bf, o_f32,
                                           N, Kpad, Npad, Fout);
  };

  run_layer(x_bf, 9, Wp1, b1, h1, nullptr, 256, 250);
  run_layer(h1,   8, Wp2, b2, h2, nullptr, 512, 500);
  run_layer(h2,   9, Wp3, b3, nullptr, (float*)d_out, 1024, 1000);
}
